// RWKV_Tmix_x070_23828478558503
// MI455X (gfx1250) — matrix operand order repaired and checked by reading, not compiled, not run on hardware
//
#include <hip/hip_runtime.h>

// ---------------- constants for this problem ----------------
#define Bz 2
#define Tz 1024
#define Cz 2048
#define Hz 32
#define Nz 64
#define Mz (Bz * Tz)   // 2048 rows
#define LDA 72         // padded LDS row stride in bf16 elements (144 B)

typedef __bf16 bf16;
typedef unsigned int u32;
typedef __attribute__((ext_vector_type(16))) bf16  v16bf;
typedef __attribute__((ext_vector_type(8)))  bf16  v8bf;
typedef __attribute__((ext_vector_type(8)))  float v8f;
typedef __attribute__((ext_vector_type(4)))  u32   u32x4;
typedef __attribute__((ext_vector_type(8)))  int   i32x8;
typedef __attribute__((ext_vector_type(4)))  int   i32x4;

// =====================================================================
// TDM: DMA a [tile_rows x tile_k] bf16 tile (row stride K elements in
// global) into LDS at lds_byte_addr with padded row stride LDA elements.
// D# packing per CDNA5 ISA 8.3/8.4. 2-D tensor -> groups 2/3 zero.
// 6-arg builtin form (amdgpu-toolchain / clang-23 profile).
// =====================================================================
__device__ __forceinline__ void tdm_load_tile(const bf16* gsrc, u32 lds_byte_addr,
                                              int K, int rows_total,
                                              int tile_k, int tile_rows) {
  unsigned long long ga = (unsigned long long)(uintptr_t)gsrc;
  u32x4 g0;
  g0[0] = 1u;                                         // count=1
  g0[1] = lds_byte_addr;                              // lds_addr
  g0[2] = (u32)(ga & 0xffffffffull);                  // global_addr[31:0]
  g0[3] = (u32)((ga >> 32) & 0x01ffffffull) | (2u << 30); // addr[56:32] | type=2

  // padding: rows of tile_k elems (tile_k/2 DWORDs) padded up to LDA elems
  const u32 pic = (tile_k == 64) ? 4u : 3u;           // 32 or 16 DWORD interval
  const u32 pac = (u32)((LDA - tile_k) / 2 - 1);      // pad DWORDs - 1
  i32x8 g1;
  g1[0] = (int)((1u << 16) | (1u << 20) | (pic << 22) | (pac << 25)); // data_size=2B, pad_enable
  g1[1] = (int)(((u32)K & 0xffffu) << 16);            // tensor_dim0[15:0]
  g1[2] = (int)((((u32)K >> 16) & 0xffffu) | (((u32)rows_total & 0xffffu) << 16));
  g1[3] = (int)(((((u32)rows_total) >> 16) & 0xffffu) | ((u32)tile_k << 16)); // tile_dim0
  g1[4] = (int)((u32)tile_rows);                      // tile_dim1 (tile_dim2=0)
  g1[5] = (int)((u32)K);                              // tensor_dim0_stride[31:0]
  g1[6] = 0;
  g1[7] = 0;
  i32x4 z4 = {0, 0, 0, 0};
  i32x8 z8 = {0, 0, 0, 0, 0, 0, 0, 0};
  __builtin_amdgcn_tensor_load_to_lds(g0, g1, z4, z4, z8, 0);
}

// =====================================================================
// bf16 WMMA GEMM with TDM-staged, double-buffered LDS tiles.
//   Cm[M,N] = A[M,K] * Bw[N,K]^T   (f32 accum)
// block = 128 threads = 4 waves; block tile 64(M) x 64(N); K-block 64.
// Wave 0 drives the TDM; per K-block each wave does 2 K-steps of
// 4 WMMAs (16x64 output) reading fragments from padded LDS.
// =====================================================================
__global__ __launch_bounds__(128)
void gemm_bf16_wmma(const bf16* __restrict__ A, const bf16* __restrict__ Bw,
                    float* __restrict__ Cm, int M, int N, int K) {
  const int lane  = threadIdx.x & 31;
  const int wave  = threadIdx.x >> 5;
  const int tileM = blockIdx.x * 64 + wave * 16;
  const int tileN = blockIdx.y * 64;
  const int row0  = blockIdx.x * 64;
  const int hi    = (lane >> 4) * 8;     // A fragment K split
  const int bhi   = (lane >> 4) * 8;     // B fragment K split

  __shared__ __attribute__((aligned(32))) bf16 shA[2][64 * LDA];
  __shared__ __attribute__((aligned(32))) bf16 shB[2][64 * LDA];

  v8f acc0 = {}, acc1 = {}, acc2 = {}, acc3 = {};

  const bool issuer = (threadIdx.x < 32);    // wave 0 (wave-uniform branch)
  const int tk0 = (K < 64) ? K : 64;
  if (issuer) {
    tdm_load_tile(A  + (size_t)row0  * K, (u32)(uintptr_t)&shA[0][0], K, M, tk0, 64);
    tdm_load_tile(Bw + (size_t)tileN * K, (u32)(uintptr_t)&shB[0][0], K, N, tk0, 64);
  }

  int buf = 0;
  for (int kt = 0; kt < K; kt += 64) {
    const int ktn = kt + 64;
    if (issuer) {
      if (ktn < K) {
        const int tkn = (K - ktn < 64) ? (K - ktn) : 64;
        tdm_load_tile(A  + (size_t)row0  * K + ktn, (u32)(uintptr_t)&shA[buf ^ 1][0], K, M, tkn, 64);
        tdm_load_tile(Bw + (size_t)tileN * K + ktn, (u32)(uintptr_t)&shB[buf ^ 1][0], K, N, tkn, 64);
        __builtin_amdgcn_s_wait_tensorcnt(2);   // current pair landed
      } else {
        __builtin_amdgcn_s_wait_tensorcnt(0);
      }
    }
    __syncthreads();

    const int tk = (K - kt < 64) ? (K - kt) : 64;
    const bf16* __restrict__ Abase = &shA[buf][0];
    const bf16* __restrict__ Bbase = &shB[buf][0];
    #pragma unroll 2
    for (int ks = 0; ks < tk; ks += 32) {
      // ---- A fragment (16x32): two 16-byte LDS loads ----
      v16bf aF;
      {
        const bf16* Ar = Abase + (size_t)((wave << 4) + (lane & 15)) * LDA + ks;
        const v8bf lo = *reinterpret_cast<const v8bf*>(Ar + hi);
        const v8bf hh = *reinterpret_cast<const v8bf*>(Ar + 16 + hi);
        #pragma unroll
        for (int j = 0; j < 8; ++j) { aF[j] = lo[j]; aF[j + 8] = hh[j]; }
      }
      // ---- 4 B fragments (32x16 each), reuse A ----
      #pragma unroll
      for (int nt = 0; nt < 4; ++nt) {
        const bf16* Br = Bbase + (size_t)((lane & 15) + nt * 16) * LDA + ks + bhi;
        const v8bf l0 = *reinterpret_cast<const v8bf*>(Br);
        const v8bf l1 = *reinterpret_cast<const v8bf*>(Br + 16);
        v16bf bF;
        #pragma unroll
        for (int j = 0; j < 8; ++j) { bF[j] = l0[j]; bF[j + 8] = l1[j]; }
        v8f* acc = (nt == 0) ? &acc0 : (nt == 1) ? &acc1 : (nt == 2) ? &acc2 : &acc3;
        *acc = __builtin_amdgcn_wmma_f32_16x16x32_bf16(false, aF, false, bF, (short)0, *acc, false, false);
      }
    }
    __syncthreads();
    buf ^= 1;
  }

  // ---- store: C/D layout: lane n=lane&15, VGPR i -> row i + 8*(lane>=16) ----
  const int mbase = tileM + ((lane >> 4) ? 8 : 0);
  const int col0  = tileN + (lane & 15);
  #pragma unroll
  for (int i = 0; i < 8; ++i) {
    const size_t r = (size_t)(mbase + i) * N;
    Cm[r + col0     ] = acc0[i];
    Cm[r + col0 + 16] = acc1[i];
    Cm[r + col0 + 32] = acc2[i];
    Cm[r + col0 + 48] = acc3[i];
  }
}

// =====================================================================
// staging / elementwise kernels
// =====================================================================
__global__ void f32_to_bf16_k(const float* __restrict__ s, bf16* __restrict__ d, size_t n) {
  size_t i = (size_t)blockIdx.x * blockDim.x + threadIdx.x;
  if (i < n) d[i] = (bf16)s[i];
}

// src [R, Cc] f32 -> dst [Cc, R] bf16 (transposed)
__global__ void f32_to_bf16_T_k(const float* __restrict__ s, bf16* __restrict__ d, int R, int Cc) {
  size_t i = (size_t)blockIdx.x * blockDim.x + threadIdx.x;
  size_t n = (size_t)R * Cc;
  if (i >= n) return;
  int r = (int)(i / Cc), c = (int)(i % Cc);
  d[(size_t)c * R + r] = (bf16)s[i];
}

// m128 [M,128] -> Abf [M,32] (columns f*32..f*32+31), bf16
__global__ void slice32_to_bf16_k(const float* __restrict__ s, bf16* __restrict__ d, int f) {
  size_t i = (size_t)blockIdx.x * blockDim.x + threadIdx.x;
  if (i >= (size_t)Mz * 32) return;
  int m = (int)(i >> 5), j = (int)(i & 31);
  d[i] = (bf16)s[(size_t)m * 128 + f * 32 + j];
}

__global__ void tanh_ip_k(float* __restrict__ x, size_t n) {
  size_t i = (size_t)blockIdx.x * blockDim.x + threadIdx.x;
  if (i < n) x[i] = tanhf(x[i]);
}

// xx = shift(x) - x ; Abf = bf16(x + xx * time_maa_x)
__global__ void shift_mix_k(const float* __restrict__ x, const float* __restrict__ shift,
                            const float* __restrict__ maa_x,
                            float* __restrict__ xx, bf16* __restrict__ Abf) {
  size_t i = (size_t)blockIdx.x * blockDim.x + threadIdx.x;
  if (i >= (size_t)Mz * Cz) return;
  int m = (int)(i / Cz), c = (int)(i % Cz);
  int b = m / Tz, t = m % Tz;
  float xc = x[i];
  float prev = (t == 0) ? shift[(size_t)b * Cz + c] : x[i - Cz];
  float xv = prev - xc;
  xx[i] = xv;
  Abf[i] = (bf16)(xc + xv * maa_x[c]);
}

// dst = x + xx * (maa[c] + mtmp)
__global__ void mix_k(const float* __restrict__ x, const float* __restrict__ xx,
                      const float* __restrict__ maa, const float* __restrict__ mtmp,
                      float* __restrict__ dst) {
  size_t i = (size_t)blockIdx.x * blockDim.x + threadIdx.x;
  if (i >= (size_t)Mz * Cz) return;
  int c = (int)(i % Cz);
  dst[i] = x[i] + xx[i] * (maa[c] + mtmp[i]);
}

// w = -softplus(-(time_decay + proj)) - 0.5
__global__ void decay_fin_k(const float* __restrict__ td, const float* __restrict__ proj,
                            float* __restrict__ w) {
  size_t i = (size_t)blockIdx.x * blockDim.x + threadIdx.x;
  if (i >= (size_t)Mz * Cz) return;
  int c = (int)(i % Cz);
  float z = td[c] + proj[i];
  w[i] = -log1pf(expf(-z)) - 0.5f;
}

// t16[m,j] = sum_c X[m,c] * W1[c,j]   (W1 is [C,16])
__global__ __launch_bounds__(64)
void lora16_in_k(const float* __restrict__ X, const float* __restrict__ W1,
                 float* __restrict__ t16) {
  int m = blockIdx.x * 4 + (threadIdx.x >> 4);
  int j = threadIdx.x & 15;
  float s = 0.f;
  const float* xr = X + (size_t)m * Cz;
  for (int c = 0; c < Cz; ++c) s += xr[c] * W1[c * 16 + j];
  t16[(size_t)m * 16 + j] = s;
}

// kk = k + t16tanh @ kkk_w2   (t16 already tanh'ed)
__global__ void kk_make_k(const float* __restrict__ k, const float* __restrict__ t16,
                          const float* __restrict__ W2, float* __restrict__ kk) {
  size_t i = (size_t)blockIdx.x * blockDim.x + threadIdx.x;
  if (i >= (size_t)Mz * Cz) return;
  int m = (int)(i / Cz), c = (int)(i % Cz);
  const float* tr = t16 + (size_t)m * 16;
  float s = 0.f;
  #pragma unroll
  for (int j = 0; j < 16; ++j) s += tr[j] * W2[(size_t)j * Cz + c];
  kk[i] = k[i] + s;
}

// per (m,h): kk /= max(||kk||, 1e-12)
__global__ void kk_norm_k(float* __restrict__ kk) {
  size_t i = (size_t)blockIdx.x * blockDim.x + threadIdx.x;
  if (i >= (size_t)Mz * Hz) return;
  int m = (int)(i / Hz), h = (int)(i % Hz);
  float* p = kk + (size_t)m * Cz + h * Nz;
  float ss = 0.f;
  #pragma unroll 4
  for (int n = 0; n < Nz; ++n) { float d = p[n]; ss += d * d; }
  float inv = 1.f / fmaxf(sqrtf(ss), 1e-12f);
  #pragma unroll 4
  for (int n = 0; n < Nz; ++n) p[n] *= inv;
}

// a = sigmoid(bias[c] + t16 @ W2)
__global__ void sig16_k(const float* __restrict__ t16, const float* __restrict__ W2,
                        const float* __restrict__ bias, float* __restrict__ a) {
  size_t i = (size_t)blockIdx.x * blockDim.x + threadIdx.x;
  if (i >= (size_t)Mz * Cz) return;
  int m = (int)(i / Cz), c = (int)(i % Cz);
  const float* tr = t16 + (size_t)m * 16;
  float s = bias[c];
  #pragma unroll
  for (int j = 0; j < 16; ++j) s += tr[j] * W2[(size_t)j * Cz + c];
  a[i] = 1.f / (1.f + expf(-s));
}

// k = (k*ma + k*a*(1-ma)) * exp(min(w*mk, 0))
__global__ void k_update_k(const float* __restrict__ t16ma, const float* __restrict__ maW2,
                           const float* __restrict__ misc_a,
                           const float* __restrict__ t16mk, const float* __restrict__ mkW2,
                           const float* __restrict__ misc_k,
                           const float* __restrict__ a, const float* __restrict__ w,
                           float* __restrict__ k) {
  size_t i = (size_t)blockIdx.x * blockDim.x + threadIdx.x;
  if (i >= (size_t)Mz * Cz) return;
  int m = (int)(i / Cz), c = (int)(i % Cz);
  const float* ta = t16ma + (size_t)m * 16;
  const float* tk = t16mk + (size_t)m * 16;
  float sa = misc_a[c], sk = misc_k[c];
  #pragma unroll
  for (int j = 0; j < 16; ++j) {
    sa += ta[j] * maW2[(size_t)j * Cz + c];
    sk += tk[j] * mkW2[(size_t)j * Cz + c];
  }
  float ma = 1.f / (1.f + expf(-sa));
  float mk = 1.f / (1.f + expf(-sk));
  float kv = k[i];
  float av = a[i];
  kv = kv * ma + kv * av * (1.f - ma);
  kv *= expf(fminf(w[i] * mk, 0.f));
  k[i] = kv;
}

// =====================================================================
// RWKV-7 recurrence: one block per (b,h); thread v owns S[:,v] in regs.
// =====================================================================
__global__ __launch_bounds__(64)
void rwkv7_scan_k(const float* __restrict__ r, const float* __restrict__ k,
                  const float* __restrict__ v, const float* __restrict__ w,
                  const float* __restrict__ kk, const float* __restrict__ a,
                  const float* __restrict__ S0, float* __restrict__ o,
                  float* __restrict__ Sout) {
  const int bh = blockIdx.x;              // b*H + h
  const int b = bh / Hz, h = bh % Hz;
  const int vi = threadIdx.x;             // 0..63
  float S[Nz];
  #pragma unroll
  for (int kx = 0; kx < Nz; ++kx) S[kx] = S0[((size_t)bh * Nz + kx) * Nz + vi];

  __shared__ float lr[Nz], lk[Nz], lw[Nz], la[Nz], lb[Nz];

  for (int t = 0; t < Tz; ++t) {
    const size_t base = (size_t)(b * Tz + t) * Cz + h * Nz;
    const float kkv = kk[base + vi];
    lr[vi] = r[base + vi];
    lk[vi] = k[base + vi];
    lw[vi] = expf(w[base + vi]);
    la[vi] = -kkv;
    lb[vi] = kkv * a[base + vi];
    __syncthreads();
    const float vt = v[base + vi];
    float sa = 0.f;
    #pragma unroll
    for (int kx = 0; kx < Nz; ++kx) sa += la[kx] * S[kx];
    float ov = 0.f;
    #pragma unroll
    for (int kx = 0; kx < Nz; ++kx) {
      S[kx] = S[kx] * lw[kx] + lk[kx] * vt + lb[kx] * sa;
      ov += lr[kx] * S[kx];
    }
    o[base + vi] = ov;
    __syncthreads();
  }
  #pragma unroll
  for (int kx = 0; kx < Nz; ++kx) Sout[((size_t)bh * Nz + kx) * Nz + vi] = S[kx];
}

// =====================================================================
// GroupNorm(per head) + faaaa bonus + gate multiply -> y
// =====================================================================
__device__ inline float blockReduce64(float vv, float* red, int i) {
  red[i] = vv; __syncthreads();
  #pragma unroll
  for (int s = 32; s > 0; s >>= 1) {
    if (i < s) red[i] += red[i + s];
    __syncthreads();
  }
  float r0 = red[0];
  __syncthreads();
  return r0;
}

__global__ __launch_bounds__(64)
void gn_bonus_gate_k(const float* __restrict__ o, const float* __restrict__ r,
                     const float* __restrict__ k, const float* __restrict__ v,
                     const float* __restrict__ g, const float* __restrict__ faaaa,
                     const float* __restrict__ lnw, const float* __restrict__ lnb,
                     float* __restrict__ y) {
  const int mh = blockIdx.x;              // m*H + h
  const int m = mh / Hz, h = mh % Hz;
  const int i = threadIdx.x;
  const size_t base = (size_t)m * Cz + h * Nz;
  __shared__ float red[Nz];

  const float ov = o[base + i];
  const float mu = blockReduce64(ov, red, i) * (1.f / Nz);
  const float d  = ov - mu;
  const float var = blockReduce64(d * d, red, i) * (1.f / Nz);
  float yn = d * rsqrtf(var + 64e-5f);
  yn = yn * lnw[h * Nz + i] + lnb[h * Nz + i];

  const float bsum = blockReduce64(r[base + i] * k[base + i] * faaaa[h * Nz + i], red, i);
  yn += bsum * v[base + i];
  y[base + i] = yn * g[base + i];
}

__global__ void copy_last_k(const float* __restrict__ x, float* __restrict__ dst) {
  size_t i = (size_t)blockIdx.x * blockDim.x + threadIdx.x;
  if (i >= (size_t)Bz * Cz) return;
  int b = (int)(i / Cz), c = (int)(i % Cz);
  dst[i] = x[((size_t)b * Tz + (Tz - 1)) * Cz + c];
}

// =====================================================================
// host orchestration
// =====================================================================
static inline dim3 eg(size_t n, int bs = 256) { return dim3((unsigned)((n + bs - 1) / bs)); }

extern "C" void kernel_launch(void* const* d_in, const int* in_sizes, int n_in,
                              void* d_out, int out_size, void* d_ws, size_t ws_size,
                              hipStream_t stream) {
  const float* x        = (const float*)d_in[0];
  const float* shift    = (const float*)d_in[1];
  const float* wkv0     = (const float*)d_in[2];
  const float* maa_x    = (const float*)d_in[3];
  const float* maa_rg   = (const float*)d_in[4];
  const float* maa_wa   = (const float*)d_in[5];
  const float* maa_k    = (const float*)d_in[6];
  const float* maa_v    = (const float*)d_in[7];
  const float* maa_w1   = (const float*)d_in[8];
  const float* maa_w2   = (const float*)d_in[9];
  const float* tdecay   = (const float*)d_in[10];
  const float* dec_w1   = (const float*)d_in[11];
  const float* dec_w2   = (const float*)d_in[12];
  const float* faaaa    = (const float*)d_in[13];
  const float* aaaaa    = (const float*)d_in[14];
  const float* aaa_w1   = (const float*)d_in[15];
  const float* aaa_w2   = (const float*)d_in[16];
  const float* kkk_w1   = (const float*)d_in[17];
  const float* kkk_w2   = (const float*)d_in[18];
  const float* gate_w1  = (const float*)d_in[19];
  const float* gate_w2  = (const float*)d_in[20];
  const float* ma_w1    = (const float*)d_in[21];
  const float* ma_w2    = (const float*)d_in[22];
  const float* misc_a   = (const float*)d_in[23];
  const float* mk_w1    = (const float*)d_in[24];
  const float* mk_w2    = (const float*)d_in[25];
  const float* misc_k   = (const float*)d_in[26];
  const float* Wr       = (const float*)d_in[27];
  const float* Wk       = (const float*)d_in[28];
  const float* Wv       = (const float*)d_in[29];
  const float* Wo       = (const float*)d_in[30];
  const float* ln_w     = (const float*)d_in[31];
  const float* ln_b     = (const float*)d_in[32];
  (void)in_sizes; (void)n_in; (void)out_size; (void)ws_size;

  float* out = (float*)d_out;
  const size_t MC = (size_t)Mz * Cz;

  // ---- workspace carve-up (f32 region, then bf16 region) ----
  float* ws   = (float*)d_ws;
  float* xx   = ws + 0 * MC;
  float* xrg  = ws + 1 * MC;
  float* xwa  = ws + 2 * MC;
  float* xk   = ws + 3 * MC;
  float* xv   = ws + 4 * MC;
  float* rbuf = ws + 5 * MC;
  float* wbuf = ws + 6 * MC;
  float* kbuf = ws + 7 * MC;
  float* vbuf = ws + 8 * MC;
  float* gbuf = ws + 9 * MC;
  float* kkb  = ws + 10 * MC;
  float* abuf = ws + 11 * MC;
  float* obuf = ws + 12 * MC;
  float* ybuf = ws + 13 * MC;
  float* mtmp = ws + 14 * MC;
  float* m128 = ws + 15 * MC;
  float* t128 = m128 + (size_t)Mz * 128;
  float* t64  = t128 + (size_t)Mz * 128;
  float* t16a = t64  + (size_t)Mz * 64;
  float* t16b = t16a + (size_t)Mz * 16;
  float* t16c = t16b + (size_t)Mz * 16;
  float* t16d = t16c + (size_t)Mz * 16;
  float* fend = t16d + (size_t)Mz * 16;
  bf16*  Abf  = (bf16*)fend;
  bf16*  Wbf  = Abf + MC;           // up to C*C bf16

  auto gemm = [&](const bf16* A, const bf16* Bw, float* Cmat, int M, int N, int K) {
    dim3 grid(M / 64, N / 64);
    gemm_bf16_wmma<<<grid, 128, 0, stream>>>(A, Bw, Cmat, M, N, K);
  };
  auto conv = [&](const float* s, bf16* d, size_t n) {
    f32_to_bf16_k<<<eg(n), 256, 0, stream>>>(s, d, n);
  };
  auto convT = [&](const float* s, bf16* d, int R, int Cc) {
    f32_to_bf16_T_k<<<eg((size_t)R * Cc), 256, 0, stream>>>(s, d, R, Cc);
  };

  // 1) token shift + xmx (bf16 A) --------------------------------------
  shift_mix_k<<<eg(MC), 256, 0, stream>>>(x, shift, maa_x, xx, Abf);

  // 2) m = tanh(xmx @ maa_w1) ------------------------------------------
  convT(maa_w1, Wbf, Cz, 128);
  gemm(Abf, Wbf, m128, Mz, 128, Cz);
  tanh_ip_k<<<eg((size_t)Mz * 128), 256, 0, stream>>>(m128, (size_t)Mz * 128);

  // 3) per-f projection + mixes ----------------------------------------
  const float* maas[4] = {maa_rg, maa_wa, maa_k, maa_v};
  float* dsts[4] = {xrg, xwa, xk, xv};
  for (int f = 0; f < 4; ++f) {
    slice32_to_bf16_k<<<eg((size_t)Mz * 32), 256, 0, stream>>>(m128, Abf, f);
    convT(maa_w2 + (size_t)f * 32 * Cz, Wbf, 32, Cz);
    gemm(Abf, Wbf, mtmp, Mz, Cz, 32);
    mix_k<<<eg(MC), 256, 0, stream>>>(x, xx, maas[f], mtmp, dsts[f]);
  }

  // 4) r = xrg @ Wr^T ---------------------------------------------------
  conv(xrg, Abf, MC);
  conv(Wr, Wbf, (size_t)Cz * Cz);
  gemm(Abf, Wbf, rbuf, Mz, Cz, Cz);

  // 5) g = tanh(xrg @ gate_w1) @ gate_w2 -------------------------------
  convT(gate_w1, Wbf, Cz, 128);
  gemm(Abf, Wbf, t128, Mz, 128, Cz);
  tanh_ip_k<<<eg((size_t)Mz * 128), 256, 0, stream>>>(t128, (size_t)Mz * 128);
  conv(t128, Abf, (size_t)Mz * 128);
  convT(gate_w2, Wbf, 128, Cz);
  gemm(Abf, Wbf, gbuf, Mz, Cz, 128);

  // 6) w (decay) --------------------------------------------------------
  conv(xwa, Abf, MC);
  convT(dec_w1, Wbf, Cz, 64);
  gemm(Abf, Wbf, t64, Mz, 64, Cz);
  tanh_ip_k<<<eg((size_t)Mz * 64), 256, 0, stream>>>(t64, (size_t)Mz * 64);
  conv(t64, Abf, (size_t)Mz * 64);
  convT(dec_w2, Wbf, 64, Cz);
  gemm(Abf, Wbf, mtmp, Mz, Cz, 64);
  decay_fin_k<<<eg(MC), 256, 0, stream>>>(tdecay, mtmp, wbuf);
  lora16_in_k<<<Mz / 4, 64, 0, stream>>>(xwa, aaa_w1, t16a);
  lora16_in_k<<<Mz / 4, 64, 0, stream>>>(xwa, ma_w1, t16b);

  // 7) k, v -------------------------------------------------------------
  conv(xk, Abf, MC);
  conv(Wk, Wbf, (size_t)Cz * Cz);
  gemm(Abf, Wbf, kbuf, Mz, Cz, Cz);
  lora16_in_k<<<Mz / 4, 64, 0, stream>>>(xk, kkk_w1, t16c);
  lora16_in_k<<<Mz / 4, 64, 0, stream>>>(xk, mk_w1, t16d);
  conv(xv, Abf, MC);
  conv(Wv, Wbf, (size_t)Cz * Cz);
  gemm(Abf, Wbf, vbuf, Mz, Cz, Cz);

  // 8) kk = normalize_head(k + tanh(xk@kkk_w1)@kkk_w2) -----------------
  tanh_ip_k<<<eg((size_t)Mz * 16), 256, 0, stream>>>(t16c, (size_t)Mz * 16);
  kk_make_k<<<eg(MC), 256, 0, stream>>>(kbuf, t16c, kkk_w2, kkb);
  kk_norm_k<<<eg((size_t)Mz * Hz), 256, 0, stream>>>(kkb);

  // 9) a gate + k update -----------------------------------------------
  sig16_k<<<eg(MC), 256, 0, stream>>>(t16a, aaa_w2, aaaaa, abuf);
  k_update_k<<<eg(MC), 256, 0, stream>>>(t16b, ma_w2, misc_a, t16d, mk_w2, misc_k,
                                         abuf, wbuf, kbuf);

  // 10) recurrence (writes obuf and S_final into d_out tail) -----------
  float* out_xlast = out + MC;
  float* out_S     = out + MC + (size_t)Bz * Cz;
  rwkv7_scan_k<<<Bz * Hz, Nz, 0, stream>>>(rbuf, kbuf, vbuf, wbuf, kkb, abuf,
                                           wkv0, obuf, out_S);
  copy_last_k<<<eg((size_t)Bz * Cz), 256, 0, stream>>>(x, out_xlast);

  // 11) GroupNorm + bonus + gate ---------------------------------------
  gn_bonus_gate_k<<<Mz * Hz, Nz, 0, stream>>>(obuf, rbuf, kbuf, vbuf, gbuf,
                                              faaaa, ln_w, ln_b, ybuf);

  // 12) xo = y @ Wo^T  -> d_out ----------------------------------------
  conv(ybuf, Abf, MC);
  conv(Wo, Wbf, (size_t)Cz * Cz);
  gemm(Abf, Wbf, out, Mz, Cz, Cz);
}
